// NetVLAD_15788299780533
// MI455X (gfx1250) — matrix operand order repaired and checked by reading, not compiled, not run on hardware
//
#include <hip/hip_runtime.h>
#include <stdint.h>

// ---------------- problem constants ----------------
#define N_IMG 32
#define DIM   512
#define PIX   4096
#define KCL   64
#define PC    32                 // pixels per workgroup chunk
#define CHUNKS (PIX / PC)        // 128
#define XNT_STRIDE 40            // bf16 elems per row of xnT [d][p] (PC + 8 pad)
#define XNP_STRIDE 520           // bf16 elems per row of xnP [p][d] (DIM + 8 pad)
#define EPSN 1e-12f

// ---------------- vector types ----------------
typedef __attribute__((ext_vector_type(16))) __bf16   v16bf;
typedef __attribute__((ext_vector_type(8)))  float    v8f;
typedef __attribute__((ext_vector_type(8)))  unsigned v8u;
typedef __attribute__((ext_vector_type(4)))  unsigned u32x4;
typedef __attribute__((ext_vector_type(8)))  int      i32x8;
typedef __attribute__((ext_vector_type(4)))  int      i32x4;

union FragAB { v8u u; v16bf b; };

__device__ __forceinline__ unsigned bf16_1(float f) {
  unsigned a = __float_as_uint(f);
  return (a + 0x7FFFu + ((a >> 16) & 1u)) >> 16;   // RNE f32 -> bf16
}
__device__ __forceinline__ unsigned bf16_2(float lo, float hi) {
  return (bf16_1(lo) & 0xFFFFu) | (bf16_1(hi) << 16);
}
__device__ __forceinline__ v8f vzero8() {
  v8f z;
#pragma unroll
  for (int i = 0; i < 8; ++i) z[i] = 0.0f;
  return z;
}
__device__ __forceinline__ v8f wmma_bf16(v16bf a, v16bf b, v8f c) {
  return __builtin_amdgcn_wmma_f32_16x16x32_bf16(false, a, false, b,
                                                 (short)0, c, false, false);
}

// ---------------- Tensor Data Mover (gfx1250) ----------------
#ifndef __has_builtin
#define __has_builtin(x) 0
#endif
#if __has_builtin(__builtin_amdgcn_tensor_load_to_lds) && \
    __has_builtin(__builtin_amdgcn_s_wait_tensorcnt)
#define USE_TDM 1
#else
#define USE_TDM 0
#endif

#if USE_TDM
// 2D tile load: tile_d0 contiguous elements (4B each) x tile_d1 rows,
// row stride = stride0 elements.  Descriptor per CDNA5 ISA ch.8.
__device__ __forceinline__ void tdm_load_2d(unsigned lds_off, const void* gptr,
                                            unsigned tile_d0, unsigned tile_d1,
                                            unsigned tens_d0, unsigned tens_d1,
                                            unsigned long long stride0) {
  unsigned long long ga = (unsigned long long)(uintptr_t)gptr;
  u32x4 g0;
  g0[0] = 1u;                                        // count=1, user desc
  g0[1] = lds_off;                                   // lds_addr
  g0[2] = (unsigned)(ga & 0xFFFFFFFFull);            // global_addr[31:0]
  g0[3] = (unsigned)((ga >> 32) & 0x1FFFFFFull) | (2u << 30);  // addr[56:32], type=2
  i32x8 g1;
  g1[0] = (int)(2u << 16);                           // data_size = 4B
  g1[1] = (int)((tens_d0 & 0xFFFFu) << 16);          // tensor_dim0[15:0]
  g1[2] = (int)((tens_d0 >> 16) | ((tens_d1 & 0xFFFFu) << 16));
  g1[3] = (int)((tens_d1 >> 16) | (tile_d0 << 16));  // tile_dim0
  g1[4] = (int)(tile_d1 & 0xFFFFu);                  // tile_dim1, tile_dim2=0
  g1[5] = (int)(unsigned)(stride0 & 0xFFFFFFFFull);  // tensor_dim0_stride[31:0]
  g1[6] = (int)(unsigned)((stride0 >> 32) & 0xFFFFull);
  g1[7] = 0;
  i32x4 z4;
  z4[0] = z4[1] = z4[2] = z4[3] = 0;
#if defined(__clang_major__) && (__clang_major__ >= 23)
  i32x8 z8;
#pragma unroll
  for (int i = 0; i < 8; ++i) z8[i] = 0;
  __builtin_amdgcn_tensor_load_to_lds(g0, g1, z4, z4, z8, 0);
#else
  __builtin_amdgcn_tensor_load_to_lds(g0, g1, z4, z4, 0);
#endif
}
#endif

// ---------------- kernel 0: zero accumulators ----------------
__global__ void nv_init_zero(float* __restrict__ out, float* __restrict__ asum,
                             float* __restrict__ gnorm) {
  int idx = blockIdx.x * 256 + threadIdx.x;
  if (idx < N_IMG * KCL * DIM) out[idx] = 0.0f;
  if (idx < N_IMG * KCL) asum[idx] = 0.0f;
  if (idx < N_IMG) gnorm[idx] = 0.0f;
}

// ---------------- kernel 1: pre-swizzle conv_w into WMMA A-fragments ----
// frag dword index = ((f*32 + lane)*8 + j), f = kt*16 + ds.
// 16-bit A layout (16x32): lanes 0-15 row k, VGPR j<4 -> K=2j,2j+1;
// j>=4 -> K=16+2(j-4); lanes 16-31: K=8.. / K=24.. respectively.
__global__ void nv_prep_conv(const float* __restrict__ conv_w,
                             unsigned* __restrict__ frag) {
  int tid = blockIdx.x * 256 + threadIdx.x;
  if (tid >= 64 * 32 * 8) return;
  int j = tid & 7;
  int lane = (tid >> 3) & 31;
  int f = tid >> 8;
  int kt = f >> 4, ds = f & 15;
  int k = kt * 16 + (lane & 15);
  int Kb = (lane < 16) ? ((j < 4) ? 0 : 16) : ((j < 4) ? 8 : 24);
  int d = ds * 32 + Kb + 2 * (j & 3);
  frag[tid] = bf16_2(conv_w[k * DIM + d], conv_w[k * DIM + d + 1]);
}

// ---------------- kernel 2: fused normalize+logits+softmax+aggregate ----
__global__ __launch_bounds__(256, 1) void nv_main(
    const float* __restrict__ x, const uint4* __restrict__ convfrag,
    float* __restrict__ asum_g, float* __restrict__ vlad_out) {
  const int blk = blockIdx.x;
  const int n = blk >> 7;          // image
  const int chunk = blk & (CHUNKS - 1);
  const int p0 = chunk * PC;
  const float* xg = x + ((size_t)n * DIM * PIX + p0);

  // LDS:
  //  s_xnT : normalized tile bf16 [d][p], padded rows  (B operand, GEMM2)
  //  s_xnP : normalized tile bf16 [p][d], padded rows  (B operand, GEMM1)
  //  s_buf : phase 0-2 raw f32 x tile [512][32] (64KB);
  //          phase 3+  logits f32 [32][68] + a bf16 [64][40] at +32KB
  __shared__ __align__(16) unsigned short s_xnT[DIM * XNT_STRIDE];  // 40KB
  __shared__ __align__(16) unsigned short s_xnP[PC * XNP_STRIDE];   // 32.5KB
  __shared__ __align__(16) unsigned char s_buf[DIM * PC * 4];       // 64KB
  __shared__ float s_inv[PC];
  __shared__ float s_red[256];
  __shared__ float s_asum[KCL];

  float* x_raw = (float*)s_buf;
  float* logit = (float*)s_buf;                              // reuse
  unsigned short* a16 = (unsigned short*)(s_buf + 32768);    // reuse

  const int t = threadIdx.x;
  if (t < KCL) s_asum[t] = 0.0f;

  // ---- phase 0: stream x chunk into LDS (TDM if available) ----
#if USE_TDM
  if (t < 32) {
    tdm_load_2d((unsigned)(uintptr_t)x_raw, xg,
                /*tile*/ PC, DIM, /*tensor*/ PIX, DIM, /*stride*/ PIX);
    __builtin_amdgcn_s_wait_tensorcnt(0);
  }
#else
  for (int i = t; i < DIM * PC; i += 256) {
    int d = i >> 5, p = i & 31;
    x_raw[i] = xg[(size_t)d * PIX + p];
  }
#endif
  __syncthreads();

  // ---- phase 1: per-pixel channel L2 norm ----
  const int p = t & 31, dg = t >> 5;            // 8 threads per pixel
  float ss = 0.0f;
  for (int i = 0; i < 64; ++i) {
    float v = x_raw[(dg * 64 + i) * PC + p];
    ss += v * v;
  }
  s_red[t] = ss;
  __syncthreads();
  if (dg == 0) {
    float s = 0.0f;
#pragma unroll
    for (int j = 0; j < 8; ++j) s += s_red[p + 32 * j];
    s_inv[p] = 1.0f / fmaxf(sqrtf(s), EPSN);
  }
  __syncthreads();

  // ---- phase 2: write normalized bf16 tile in both layouts ----
  {
    float inv = s_inv[p];
    for (int i = 0; i < 64; ++i) {
      int d = dg * 64 + i;
      unsigned short b = (unsigned short)bf16_1(x_raw[d * PC + p] * inv);
      s_xnT[d * XNT_STRIDE + p] = b;
      s_xnP[p * XNP_STRIDE + d] = b;
    }
  }
  __syncthreads();

  // ---- phase 3: logits = conv_w @ xn  (WMMA bf16, f32 accum) ----
  const int wv = t >> 5, lane = t & 31;
  {
    const int kt = wv & 3, pt = wv >> 2;        // 4 k-tiles x 2 p-tiles
    const int pcol = pt * 16 + (lane & 15);
    const int Kb = (lane < 16) ? 0 : 8;         // B-operand K base (table order)
    const unsigned* xP = (const unsigned*)s_xnP;
    const int pbase = pcol * (XNP_STRIDE / 2) + (Kb >> 1);
    v8f acc = vzero8();
#pragma unroll
    for (int ds = 0; ds < 16; ++ds) {
      int f = kt * 16 + ds;
      uint4 q0 = convfrag[(size_t)(f * 32 + lane) * 2 + 0];
      uint4 q1 = convfrag[(size_t)(f * 32 + lane) * 2 + 1];
      FragAB A;
      A.u[0] = q0.x; A.u[1] = q0.y; A.u[2] = q0.z; A.u[3] = q0.w;
      A.u[4] = q1.x; A.u[5] = q1.y; A.u[6] = q1.z; A.u[7] = q1.w;
      FragAB B;
      int bb = pbase + ds * 16;                 // d0/2 = ds*16 dwords
#pragma unroll
      for (int j = 0; j < 4; ++j) { B.u[j] = xP[bb + j]; B.u[4 + j] = xP[bb + 8 + j]; }   // 2x ds_load_b128
      acc = wmma_bf16(A.b, B.b, acc);
    }
    // store C tile into logits[p][k] (f32)
    int kb = kt * 16 + 8 * (lane >> 4);
    int prow = pt * 16 + (lane & 15);
#pragma unroll
    for (int r = 0; r < 8; ++r) logit[prow * 68 + kb + r] = acc[r];
  }
  __syncthreads();

  // ---- phase 4: softmax over K per pixel; a -> bf16; asum accumulation ----
  {
    int pp = t >> 3;          // pixel 0..31
    int g = t & 7;            // 8 threads per pixel, 8 k's each
    float l[8];
#pragma unroll
    for (int i = 0; i < 8; ++i) l[i] = logit[pp * 68 + g * 8 + i];
    float m = l[0];
#pragma unroll
    for (int i = 1; i < 8; ++i) m = fmaxf(m, l[i]);
    for (int off = 4; off; off >>= 1) m = fmaxf(m, __shfl_xor(m, off, 32));
    float e[8], s = 0.0f;
#pragma unroll
    for (int i = 0; i < 8; ++i) { e[i] = __expf(l[i] - m); s += e[i]; }
    for (int off = 4; off; off >>= 1) s += __shfl_xor(s, off, 32);
    float invs = 1.0f / s;
#pragma unroll
    for (int i = 0; i < 8; ++i) {
      float a = e[i] * invs;
      a16[(g * 8 + i) * XNT_STRIDE + pp] = (unsigned short)bf16_1(a);
      atomicAdd(&s_asum[g * 8 + i], a);
    }
  }
  __syncthreads();
  if (t < KCL) atomicAdd(&asum_g[n * KCL + t], s_asum[t]);

  // ---- phase 5: vlad += a @ xn^T  (128 tiles, 16 per wave) ----
  {
    const int kt2 = wv & 3;
    const int dtb = (wv >> 2) * 16;             // 16 d-tiles per wave
    const unsigned* aU = (const unsigned*)a16;
    const unsigned* xU = (const unsigned*)s_xnT;
    // A fragment: a[kt2] 16x32 (whole chunk of pixels)
    int krow = kt2 * 16 + (lane & 15);
    int abase = krow * (XNT_STRIDE / 2) + ((lane < 16) ? 0 : 8);
    FragAB Af;
#pragma unroll
    for (int j = 0; j < 4; ++j) {
      Af.u[j] = aU[abase + j];
      Af.u[4 + j] = aU[abase + 4 + j];
    }
    v8f vacc[16];
#pragma unroll
    for (int dt = 0; dt < 16; ++dt) {
      int d = (dtb + dt) * 16 + (lane & 15);
      int bbase = d * (XNT_STRIDE / 2) + ((lane < 16) ? 0 : 8);
      FragAB Bf;
#pragma unroll
      for (int j = 0; j < 8; ++j) Bf.u[j] = xU[bbase + j];
      vacc[dt] = wmma_bf16(Af.b, Bf.b, vzero8());
    }
    int kb = kt2 * 16 + 8 * (lane >> 4);
#pragma unroll
    for (int dt = 0; dt < 16; ++dt) {
      int dcol = (dtb + dt) * 16 + (lane & 15);
#pragma unroll
      for (int r = 0; r < 8; ++r)
        atomicAdd(&vlad_out[((size_t)n * KCL + kb + r) * DIM + dcol],
                  vacc[dt][r]);
    }
  }
}

// ---------------- kernel 3: centroid subtract + intra-L2 norm ----------
__global__ __launch_bounds__(128) void nv_fin_a(
    float* __restrict__ vout, const float* __restrict__ asum,
    const float* __restrict__ cent, float* __restrict__ gnorm) {
  int nk = blockIdx.x;                 // n*64 + k
  int n = nk >> 6, k = nk & 63;
  float aS = asum[nk];
  size_t base = (size_t)nk * DIM;
  float y[4], ss = 0.0f;
#pragma unroll
  for (int i = 0; i < 4; ++i) {
    int d = threadIdx.x + 128 * i;
    float v = vout[base + d] - aS * cent[k * DIM + d];
    y[i] = v;
    ss += v * v;
  }
  for (int off = 16; off; off >>= 1) ss += __shfl_xor(ss, off, 32);
  __shared__ float r4[4];
  __shared__ float sinv;
  if ((threadIdx.x & 31) == 0) r4[threadIdx.x >> 5] = ss;
  __syncthreads();
  if (threadIdx.x == 0) {
    float s = r4[0] + r4[1] + r4[2] + r4[3];
    float inv = 1.0f / fmaxf(sqrtf(s), EPSN);
    sinv = inv;
    atomicAdd(&gnorm[n], s * inv * inv);   // ||row after intra-norm||^2
  }
  __syncthreads();
  float inv = sinv;
#pragma unroll
  for (int i = 0; i < 4; ++i) vout[base + threadIdx.x + 128 * i] = y[i] * inv;
}

// ---------------- kernel 4: global L2 norm per image ----------
__global__ void nv_fin_b(float* __restrict__ out,
                         const float* __restrict__ gnorm) {
  int idx = blockIdx.x * 256 + threadIdx.x;
  if (idx >= N_IMG * KCL * DIM) return;
  int n = idx >> 15;                    // K*D = 32768
  out[idx] *= 1.0f / fmaxf(sqrtf(gnorm[n]), EPSN);
}

// ---------------- launcher ----------------
extern "C" void kernel_launch(void* const* d_in, const int* in_sizes, int n_in,
                              void* d_out, int out_size, void* d_ws,
                              size_t ws_size, hipStream_t stream) {
  const float* x = (const float*)d_in[0];       // [32,512,64,64]
  const float* conv_w = (const float*)d_in[1];  // [64,512]
  const float* cent = (const float*)d_in[2];    // [64,512]
  float* out = (float*)d_out;                   // [32, 64*512]
  char* ws = (char*)d_ws;
  float* asum = (float*)ws;                     // 2048 f32
  float* gnorm = (float*)(ws + 8192);           // 32 f32
  unsigned* cfrag = (unsigned*)(ws + 8320);     // 16384 dwords (64KB)

  nv_init_zero<<<4096, 256, 0, stream>>>(out, asum, gnorm);
  nv_prep_conv<<<64, 256, 0, stream>>>(conv_w, cfrag);
  nv_main<<<N_IMG * CHUNKS, 256, 0, stream>>>(x, (const uint4*)cfrag, asum,
                                              out);
  nv_fin_a<<<N_IMG * KCL, 128, 0, stream>>>(out, asum, cent, gnorm);
  nv_fin_b<<<4096, 256, 0, stream>>>(out, gnorm);
}
